// AttentionalRNNDecoder_1906965479942
// MI455X (gfx1250) — matrix operand order repaired and checked by reading, not compiled, not run on hardware
//
#include <hip/hip_runtime.h>

typedef unsigned short u16;
typedef unsigned int   u32;
typedef __attribute__((ext_vector_type(16))) __bf16 v16bf;
typedef __attribute__((ext_vector_type(8)))  float  v8f;

#define D_MODEL 1024
#define NHEAD   8
#define DHEAD   128
#define SLEN    2048
#define TLEN    256
#define BATCH   16

// ---------- bf16 helpers (round-to-nearest-even) ----------
__device__ inline u16 f2bf(float x) {
  u32 u = __float_as_uint(x);
  u32 r = (u + 0x7FFFu + ((u >> 16) & 1u)) >> 16;
  return (u16)r;
}
__device__ inline float bf2f(u32 bits16) {
  return __uint_as_float(bits16 << 16);
}

// ---------- one 16x16 output tile of  X[16,K](bf16) @ W^T  (W is [N,K] bf16) ----------
// A fragment (16x32 bf16): lanes 0-15 -> M=lane, K in {0..7,16..23}; lanes 16-31 -> K in {8..15,24..31}
// B fragment (32x16 bf16): lanes 0-15 -> N=lane, K in {0..7,16..23}; lanes 16-31 -> K in {8..15,24..31}
__device__ inline v8f tile_mm(const u16* __restrict__ X, const u16* __restrict__ W,
                              int K, int n0, int lane, v8f acc) {
  const int m    = lane & 15;
  const int half = lane >> 4;
  const int n    = lane & 15;
  const u16* xrow = X + (size_t)m * K;
  const u16* wrow = W + (size_t)(n0 + n) * K;
#pragma unroll 4
  for (int k0 = 0; k0 < K; k0 += 32) {
    union { v16bf v; uint4 q[2]; } a, b;
    a.q[0] = *(const uint4*)(xrow + k0 + half * 8);        // K pairs {0..7}/{8..15}
    a.q[1] = *(const uint4*)(xrow + k0 + 16 + half * 8);   // K pairs {16..23}/{24..31}
    b.q[0] = *(const uint4*)(wrow + k0 + half * 8);        // K {0..7}/{8..15}
    b.q[1] = *(const uint4*)(wrow + k0 + 16 + half * 8);
    acc = __builtin_amdgcn_wmma_f32_16x16x32_bf16(false, a.v, false, b.v,
                                                  (short)0, acc, false, false);
  }
  return acc;
}

// ---------- generic gates/projection GEMM:  out[16,N] = X1@W1^T (+ X2@W2^T) + b1 (+ b2) ----------
// writes fp32 (outf) or bf16 (outb)
__global__ __launch_bounds__(256)
void gemm16_kernel(const u16* __restrict__ X1, const u16* __restrict__ X2,
                   const u16* __restrict__ W1, const u16* __restrict__ W2,
                   const float* __restrict__ b1, const float* __restrict__ b2,
                   float* __restrict__ outf, u16* __restrict__ outb, int N) {
  const int K = D_MODEL;
  int wave = (int)((blockIdx.x * blockDim.x + threadIdx.x) >> 5);
  int lane = threadIdx.x & 31;
  if (wave >= N / 16) return;
  int n0 = wave * 16;
  v8f acc = {};
  acc = tile_mm(X1, W1, K, n0, lane, acc);
  if (X2) acc = tile_mm(X2, W2, K, n0, lane, acc);
  int n = lane & 15, half = lane >> 4;
  float bias = b1[n0 + n] + (b2 ? b2[n0 + n] : 0.0f);
  if (outf) {
#pragma unroll
    for (int r = 0; r < 8; ++r)
      outf[(size_t)(r + half * 8) * N + n0 + n] = acc[r] + bias;
  } else {
#pragma unroll
    for (int r = 0; r < 8; ++r)
      outb[(size_t)(r + half * 8) * N + n0 + n] = f2bf(acc[r] + bias);
  }
}

// ---------- K/V projection of encoder memory (bf16 output, [S*B, D] == [S,B,H,dh]) ----------
__global__ __launch_bounds__(256)
void kvproj_kernel(const u16* __restrict__ featb,
                   const u16* __restrict__ Wk, const float* __restrict__ bk,
                   const u16* __restrict__ Wv, const float* __restrict__ bv,
                   u16* __restrict__ Kp, u16* __restrict__ Vp) {
  const int K = D_MODEL;
  const int NT = D_MODEL / 16;                 // 64 n-tiles
  const int TOT = (SLEN * BATCH / 16) * NT;    // 131072 tiles per output
  int id = (int)((blockIdx.x * blockDim.x + threadIdx.x) >> 5);
  int lane = threadIdx.x & 31;
  int sel = id >= TOT;
  if (sel) id -= TOT;
  if (id >= TOT) return;
  int mt = id / NT, nt = id % NT;
  const u16* X = featb + (size_t)mt * 16 * K;
  const u16* W = sel ? Wv : Wk;
  const float* bias = sel ? bv : bk;
  u16* O = sel ? Vp : Kp;
  v8f acc = {};
  acc = tile_mm(X, W, K, nt * 16, lane, acc);
  int n = lane & 15, half = lane >> 4;
  float bb = bias[nt * 16 + n];
#pragma unroll
  for (int r = 0; r < 8; ++r)
    O[(size_t)(mt * 16 + r + half * 8) * D_MODEL + nt * 16 + n] = f2bf(acc[r] + bb);
}

// ---------- LSTM elementwise state update (h out in bf16, c out in fp32) ----------
__global__ __launch_bounds__(256)
void lstm_update_kernel(const float* __restrict__ g, const float* __restrict__ cprev,
                        u16* __restrict__ hout, float* __restrict__ cout,
                        float* __restrict__ yout) {
  int idx = blockIdx.x * blockDim.x + threadIdx.x;   // 0..16383
  if (idx >= BATCH * D_MODEL) return;
  int b = idx >> 10, d = idx & 1023;
  const float* gr = g + (size_t)b * 4 * D_MODEL;
  float gi = gr[d], gf = gr[D_MODEL + d], gg = gr[2 * D_MODEL + d], go = gr[3 * D_MODEL + d];
  float si = 1.0f / (1.0f + expf(-gi));
  float sf = 1.0f / (1.0f + expf(-gf));
  float so = 1.0f / (1.0f + expf(-go));
  float cn = sf * cprev[idx] + si * tanhf(gg);
  float hn = so * tanhf(cn);
  cout[idx] = cn;
  hout[idx] = f2bf(hn);
  if (yout) yout[idx] = hn;
}

// ---------- per-(b,h) cross-attention: scores -> softmax -> context (bf16 out) ----------
__global__ __launch_bounds__(256)
void attn_kernel(const float* __restrict__ qbuf, const u16* __restrict__ Kp,
                 const u16* __restrict__ Vp, u16* __restrict__ ctxb) {
  __shared__ float sc[SLEN];
  __shared__ float qs[DHEAD];
  __shared__ float red[256];
  __shared__ float cred[DHEAD];
  const float scale = 0.08838834764831845f;   // 1/sqrt(128)
  int bh = blockIdx.x;
  int b = bh >> 3, h = bh & 7;
  int tid = threadIdx.x;
  if (tid < DHEAD) qs[tid] = qbuf[(size_t)b * D_MODEL + h * DHEAD + tid];
  __syncthreads();

  const u16* Kb = Kp + ((size_t)b * NHEAD + h) * DHEAD;   // + s * B*H*dh
  const size_t srow = (size_t)BATCH * NHEAD * DHEAD;      // 16384 elems per s
  float lmax = -3.4e38f;
#pragma unroll
  for (int i = 0; i < SLEN / 256; ++i) {
    int s = tid + i * 256;
    const u32* kr = (const u32*)(Kb + (size_t)s * srow);
    float acc = 0.0f;
#pragma unroll 8
    for (int j = 0; j < DHEAD / 2; ++j) {
      u32 w = kr[j];
      acc += qs[2 * j] * bf2f(w & 0xFFFFu) + qs[2 * j + 1] * bf2f(w >> 16);
    }
    acc *= scale;
    sc[s] = acc;
    lmax = fmaxf(lmax, acc);
  }
  // block max
  red[tid] = lmax; __syncthreads();
  for (int off = 128; off > 0; off >>= 1) {
    if (tid < off) red[tid] = fmaxf(red[tid], red[tid + off]);
    __syncthreads();
  }
  float mx = red[0]; __syncthreads();
  // exp + block sum
  float lsum = 0.0f;
#pragma unroll
  for (int i = 0; i < SLEN / 256; ++i) {
    int s = tid + i * 256;
    float e = expf(sc[s] - mx);
    sc[s] = e;
    lsum += e;
  }
  red[tid] = lsum; __syncthreads();
  for (int off = 128; off > 0; off >>= 1) {
    if (tid < off) red[tid] += red[tid + off];
    __syncthreads();
  }
  float ssum = red[0]; __syncthreads();
  // context: 2 partial s-strips per output element
  int j = tid & (DHEAD - 1), part = tid >> 7;
  const u16* Vb = Vp + ((size_t)b * NHEAD + h) * DHEAD + j;
  float acc = 0.0f;
  for (int s = part; s < SLEN; s += 2)
    acc += sc[s] * bf2f((u32)Vb[(size_t)s * srow]);
  if (part) cred[j] = acc;
  __syncthreads();
  if (!part)
    ctxb[(size_t)b * D_MODEL + h * DHEAD + j] = f2bf((acc + cred[j]) / ssum);
}

// ---------- misc ----------
__global__ __launch_bounds__(256)
void cvt_bf16_kernel(const float* __restrict__ s, u16* __restrict__ d, int n) {
  int i = blockIdx.x * blockDim.x + threadIdx.x;
  if (i < n) d[i] = f2bf(s[i]);
}
__global__ __launch_bounds__(256)
void zero_state_kernel(u16* hb, float* c, int n) {
  int i = blockIdx.x * blockDim.x + threadIdx.x;
  if (i < n) { hb[i] = 0; c[i] = 0.0f; }
}

extern "C" void kernel_launch(void* const* d_in, const int* in_sizes, int n_in,
                              void* d_out, int out_size, void* d_ws, size_t ws_size,
                              hipStream_t stream) {
  const float* features = (const float*)d_in[0];
  const float* prev     = (const float*)d_in[1];
  const float* W_ih1 = (const float*)d_in[2];
  const float* W_hh1 = (const float*)d_in[3];
  const float* b_ih1 = (const float*)d_in[4];
  const float* b_hh1 = (const float*)d_in[5];
  const float* W_ih2 = (const float*)d_in[6];
  const float* W_hh2 = (const float*)d_in[7];
  const float* b_ih2 = (const float*)d_in[8];
  const float* b_hh2 = (const float*)d_in[9];
  const float* Wq = (const float*)d_in[10]; const float* bq = (const float*)d_in[11];
  const float* Wk = (const float*)d_in[12]; const float* bk = (const float*)d_in[13];
  const float* Wv = (const float*)d_in[14]; const float* bv = (const float*)d_in[15];
  const float* Wo = (const float*)d_in[16]; const float* bo = (const float*)d_in[17];
  float* out = (float*)d_out;

  // -------- workspace layout --------
  char* p = (char*)d_ws;
  auto alloc = [&](size_t bytes) -> char* {
    char* r = p; p += (bytes + 255) & ~(size_t)255; return r;
  };
  const int WBIG = 4 * D_MODEL * D_MODEL;       // 4,194,304
  const int WSML = D_MODEL * D_MODEL;           // 1,048,576
  const int NFEAT = SLEN * BATCH * D_MODEL;     // 33,554,432
  const int NPREV = TLEN * BATCH * D_MODEL;     //  4,194,304
  u16* Wih1b = (u16*)alloc((size_t)WBIG * 2);
  u16* Whh1b = (u16*)alloc((size_t)WBIG * 2);
  u16* Wih2b = (u16*)alloc((size_t)WBIG * 2);
  u16* Whh2b = (u16*)alloc((size_t)WBIG * 2);
  u16* Wqb = (u16*)alloc((size_t)WSML * 2);
  u16* Wkb = (u16*)alloc((size_t)WSML * 2);
  u16* Wvb = (u16*)alloc((size_t)WSML * 2);
  u16* Wob = (u16*)alloc((size_t)WSML * 2);
  u16* Kp    = (u16*)alloc((size_t)NFEAT * 2);
  u16* Vp    = (u16*)alloc((size_t)NFEAT * 2);
  u16* featb = (u16*)alloc((size_t)NFEAT * 2);
  u16* prevb = (u16*)alloc((size_t)NPREV * 2);
  float* g1 = (float*)alloc((size_t)BATCH * 4 * D_MODEL * 4);
  float* g2 = (float*)alloc((size_t)BATCH * 4 * D_MODEL * 4);
  const int ELEM = BATCH * D_MODEL;             // 16384
  u16*   hAb   = (u16*)alloc((size_t)ELEM * 2);
  u16*   h1b   = (u16*)alloc((size_t)ELEM * 2);
  u16*   ctxb  = (u16*)alloc((size_t)ELEM * 2);
  u16*   attnb = (u16*)alloc((size_t)ELEM * 2);
  float* cA    = (float*)alloc((size_t)ELEM * 4);
  float* c1    = (float*)alloc((size_t)ELEM * 4);
  float* qbuf  = (float*)alloc((size_t)ELEM * 4);

  // -------- one-time (per launch) precompute --------
  cvt_bf16_kernel<<<WBIG / 256, 256, 0, stream>>>(W_ih1, Wih1b, WBIG);
  cvt_bf16_kernel<<<WBIG / 256, 256, 0, stream>>>(W_hh1, Whh1b, WBIG);
  cvt_bf16_kernel<<<WBIG / 256, 256, 0, stream>>>(W_ih2, Wih2b, WBIG);
  cvt_bf16_kernel<<<WBIG / 256, 256, 0, stream>>>(W_hh2, Whh2b, WBIG);
  cvt_bf16_kernel<<<WSML / 256, 256, 0, stream>>>(Wq, Wqb, WSML);
  cvt_bf16_kernel<<<WSML / 256, 256, 0, stream>>>(Wk, Wkb, WSML);
  cvt_bf16_kernel<<<WSML / 256, 256, 0, stream>>>(Wv, Wvb, WSML);
  cvt_bf16_kernel<<<WSML / 256, 256, 0, stream>>>(Wo, Wob, WSML);
  cvt_bf16_kernel<<<NFEAT / 256, 256, 0, stream>>>(features, featb, NFEAT);
  cvt_bf16_kernel<<<NPREV / 256, 256, 0, stream>>>(prev, prevb, NPREV);
  zero_state_kernel<<<(ELEM + 255) / 256, 256, 0, stream>>>(hAb, cA, ELEM);
  // K/V projection: 2 * 131072 wave-tiles, 8 waves per block
  kvproj_kernel<<<32768, 256, 0, stream>>>(featb, Wkb, bk, Wvb, bv, Kp, Vp);

  // -------- sequential decode --------
  for (int t = 0; t < TLEN; ++t) {
    const u16* x_t = prevb + (size_t)t * ELEM;
    // LSTM1 gates:  x_t@Wih1^T + hA@Whh1^T + biases   -> g1[16,4096] (fp32)
    gemm16_kernel<<<32, 256, 0, stream>>>(x_t, hAb, Wih1b, Whh1b, b_ih1, b_hh1,
                                          g1, nullptr, 4 * D_MODEL);
    lstm_update_kernel<<<64, 256, 0, stream>>>(g1, cA, h1b, c1, nullptr);
    // q projection -> qbuf[16,1024] (fp32)
    gemm16_kernel<<<8, 256, 0, stream>>>(h1b, nullptr, Wqb, nullptr, bq, nullptr,
                                         qbuf, nullptr, D_MODEL);
    // cross-attention over L2-resident bf16 K/V -> ctx (bf16)
    attn_kernel<<<BATCH * NHEAD, 256, 0, stream>>>(qbuf, Kp, Vp, ctxb);
    // output projection -> attn[16,1024] (bf16, feeds LSTM2 A-operand)
    gemm16_kernel<<<8, 256, 0, stream>>>(ctxb, nullptr, Wob, nullptr, bo, nullptr,
                                         nullptr, attnb, D_MODEL);
    // LSTM2 gates: attn@Wih2^T + h1@Whh2^T            -> g2[16,4096] (fp32)
    gemm16_kernel<<<32, 256, 0, stream>>>(attnb, h1b, Wih2b, Whh2b, b_ih2, b_hh2,
                                          g2, nullptr, 4 * D_MODEL);
    // h2,c2 -> become next step's (hA,cA); h2 is the step output
    lstm_update_kernel<<<64, 256, 0, stream>>>(g2, c1, hAb, cA, out + (size_t)t * ELEM);
  }
}
